// FullConvolution_14877766713723
// MI455X (gfx1250) — compile-verified, matrix operand order checked by reading, not run on hardware
//
#include <hip/hip_runtime.h>

// CDNA5 / gfx1250 rulebook sparse convolution:
//   async gather (GLOBAL_LOAD_ASYNC_TO_LDS_B128, double-buffered; weights staged
//   through the same async path) -> per-offset GEMM (V_WMMA_F32_16X16X32_BF16,
//   bf16 hi/lo split ~ f32 accuracy) -> scatter-add (global_atomic_add_f32).
//
// features : [N_IN, 64] f32        d_in[0]
// weight   : [K, 64, 64] f32       d_in[1]
// bias     : [64] f32              d_in[2]
// rules_in : [K, N_IN] i32         d_in[3]
// rules_out: [K, N_IN] i32         d_in[4]
// out      : [N_OUT, 64] f32       d_out

typedef __attribute__((ext_vector_type(16))) __bf16 v16bf;
typedef __attribute__((ext_vector_type(8)))  float  v8f;
typedef __attribute__((ext_vector_type(4)))  float  v4f;

#define C_DIM 64
#define CHUNK_ROWS 32
#define CHUNKS_PER_BLOCK 8
#define ROWS_PER_BLOCK (CHUNK_ROWS * CHUNKS_PER_BLOCK)
#define LDS_STRIDE 68   // 64 + 4 pad floats: 16B-aligned rows, avoids bank conflicts

// K index inside a 16-bit A/B WMMA fragment (wave32, 16x16x32):
// lanes 0-15 : VGPR0-3 -> K=0..7,  VGPR4-7 -> K=16..23
// lanes 16-31: VGPR0-3 -> K=8..15, VGPR4-7 -> K=24..31
// => fragment elems [0..7] are K = half*8 + 0..7, elems [8..15] are K = half*8 + 16..23.
__device__ __forceinline__ int kmap(int i, int half) {
    return i + 8 * ((i >> 3) + half);
}

// memory -> LDS async copy, 16 bytes, tracked by ASYNCcnt (no VGPR round-trip)
__device__ __forceinline__ void async_copy_b128(uint32_t lds_addr, const void* gaddr) {
    asm volatile("global_load_async_to_lds_b128 %0, %1, off"
                 :: "v"(lds_addr), "v"((uint64_t)(uintptr_t)gaddr) : "memory");
}
__device__ __forceinline__ void wait_asynccnt0() {
    asm volatile("s_wait_asynccnt 0x0" ::: "memory");
}
// guaranteed single-instruction fp32 scatter-add
__device__ __forceinline__ void atomic_add_f32(float* p, float v) {
    asm volatile("global_atomic_add_f32 %0, %1, off"
                 :: "v"((uint64_t)(uintptr_t)p), "v"(v) : "memory");
}

__global__ __launch_bounds__(256) void bias_init_kernel(
    float* __restrict__ out, const float* __restrict__ bias, long long n4)
{
    long long i = (long long)blockIdx.x * blockDim.x + threadIdx.x;
    if (i < n4) {
        int c = (int)((i * 4) & (C_DIM - 1));   // 4 | 64 -> aligned & periodic
        float4 b = *(const float4*)(bias + c);
        *(float4*)(out + i * 4) = b;
    }
}

__global__ __launch_bounds__(256) void sparse_conv_wmma_kernel(
    const float* __restrict__ features,
    const float* __restrict__ weight,
    const int*   __restrict__ rules_in,
    const int*   __restrict__ rules_out,
    float*       __restrict__ out,
    int n_in)
{
    __shared__ float B_lds[C_DIM * LDS_STRIDE];            // weight[k] 64x64 (padded)
    __shared__ float A_lds[2][CHUNK_ROWS * LDS_STRIDE];    // double-buffered gather tile
    __shared__ int   out_idx[2][CHUNK_ROWS];

    const int tid      = threadIdx.x;
    const int lane     = tid & 31;
    const int wave     = tid >> 5;        // 0..7
    const int half     = lane >> 4;       // 0 or 1 (lane group)
    const int ln16     = lane & 15;
    const int row_tile = wave >> 2;       // 0..1 : 16-row tile within 32-row chunk
    const int col_base = (wave & 3) * 16; // 4 col tiles of 16

    const int k = blockIdx.y;             // filter offset
    const float* Wk  = weight    + (size_t)k * (C_DIM * C_DIM);
    const int*   rin = rules_in  + (size_t)k * n_in;
    const int*   rou = rules_out + (size_t)k * n_in;

    // ---- stage weight[k] into LDS via the async engine (no VGPR round-trip) ----
    for (int t = tid; t < (C_DIM * C_DIM) / 4; t += blockDim.x) {
        int r  = t >> 4;            // 0..63
        int c4 = (t & 15) << 2;     // 0,4,..,60
        async_copy_b128((uint32_t)(uintptr_t)&B_lds[r * LDS_STRIDE + c4],
                        Wk + r * C_DIM + c4);
    }
    wait_asynccnt0();
    __syncthreads();

    // ---- per-wave B fragments in registers (block-invariant), bf16 hi/lo split ----
    v16bf b_hi[2], b_lo[2];
#pragma unroll
    for (int kc = 0; kc < 2; ++kc) {
#pragma unroll
        for (int i = 0; i < 16; ++i) {
            int kk = kc * 32 + kmap(i, half);
            float f = B_lds[kk * LDS_STRIDE + col_base + ln16];
            __bf16 h = (__bf16)f;
            b_hi[kc][i] = h;
            b_lo[kc][i] = (__bf16)(f - (float)h);
        }
    }

    const int block_base = blockIdx.x * ROWS_PER_BLOCK;
    const int nch = min(CHUNKS_PER_BLOCK,
                        (n_in - block_base + CHUNK_ROWS - 1) / CHUNK_ROWS);

    // issue async gather of chunk `ch` into buffer ch&1 (+ stage its out-indices)
    auto issue_chunk = [&](int ch) {
        const int base = block_base + ch * CHUNK_ROWS;
        const int rows = min(CHUNK_ROWS, n_in - base);
        float* Abuf = A_lds[ch & 1];
        // 512 float4 tasks / 256 threads: thread handles rows r0 and r0+16.
        const int r0 = tid >> 4;            // 0..15
        const int c4 = (tid & 15) << 2;     // 0,4,..,60
        const int r1 = r0 + 16;             // 16..31
        int s0 = (r0 < rows) ? rin[base + r0] : -1;
        int s1 = (r1 < rows) ? rin[base + r1] : -1;
        if (s0 >= 0)
            async_copy_b128((uint32_t)(uintptr_t)&Abuf[r0 * LDS_STRIDE + c4],
                            features + (size_t)s0 * C_DIM + c4);
        if (s1 >= 0)
            async_copy_b128((uint32_t)(uintptr_t)&Abuf[r1 * LDS_STRIDE + c4],
                            features + (size_t)s1 * C_DIM + c4);
        if (tid < CHUNK_ROWS)
            out_idx[ch & 1][tid] = (tid < rows) ? rou[base + tid] : -1;
    };

    if (nch > 0) issue_chunk(0);

    for (int ch = 0; ch < nch; ++ch) {
        wait_asynccnt0();     // this wave's copies for chunk `ch` are done
        __syncthreads();      // all waves' copies done; prev buffer free for reuse

        if (ch + 1 < nch) issue_chunk(ch + 1);   // overlap next gather with compute

        // ---- A fragments (vectorized LDS loads) + 6 WMMAs ----
        const float* Abase =
            &A_lds[ch & 1][(row_tile * 16 + ln16) * LDS_STRIDE];
        v8f acc = {};
#pragma unroll
        for (int kc = 0; kc < 2; ++kc) {
            const float* ap = Abase + kc * 32 + half * 8;
            v4f f0 = *(const v4f*)(ap + 0);    // K = base8 + 0..3
            v4f f1 = *(const v4f*)(ap + 4);    // K = base8 + 4..7
            v4f f2 = *(const v4f*)(ap + 16);   // K = base8 + 16..19
            v4f f3 = *(const v4f*)(ap + 20);   // K = base8 + 20..23
            v16bf a_hi, a_lo;
#pragma unroll
            for (int j = 0; j < 4; ++j) {
                float x; __bf16 h;
                x = f0[j]; h = (__bf16)x; a_hi[j]      = h; a_lo[j]      = (__bf16)(x - (float)h);
                x = f1[j]; h = (__bf16)x; a_hi[4 + j]  = h; a_lo[4 + j]  = (__bf16)(x - (float)h);
                x = f2[j]; h = (__bf16)x; a_hi[8 + j]  = h; a_lo[8 + j]  = (__bf16)(x - (float)h);
                x = f3[j]; h = (__bf16)x; a_hi[12 + j] = h; a_lo[12 + j] = (__bf16)(x - (float)h);
            }
            acc = __builtin_amdgcn_wmma_f32_16x16x32_bf16(
                      false, a_hi, false, b_hi[kc], (short)0, acc, false, false);
            acc = __builtin_amdgcn_wmma_f32_16x16x32_bf16(
                      false, a_hi, false, b_lo[kc], (short)0, acc, false, false);
            acc = __builtin_amdgcn_wmma_f32_16x16x32_bf16(
                      false, a_lo, false, b_hi[kc], (short)0, acc, false, false);
        }

        // ---- scatter-add: C/D layout -> element (m = r + 8*half, n = ln16) ----
#pragma unroll
        for (int r = 0; r < 8; ++r) {
            int m   = r + 8 * half;
            int dst = out_idx[ch & 1][row_tile * 16 + m];
            if (dst >= 0)
                atomic_add_f32(&out[(size_t)dst * C_DIM + col_base + ln16], acc[r]);
        }
    }
}

extern "C" void kernel_launch(void* const* d_in, const int* in_sizes, int n_in_args,
                              void* d_out, int out_size, void* d_ws, size_t ws_size,
                              hipStream_t stream) {
    const float* features  = (const float*)d_in[0];
    const float* weight    = (const float*)d_in[1];
    const float* bias      = (const float*)d_in[2];
    const int*   rules_in  = (const int*)d_in[3];
    const int*   rules_out = (const int*)d_in[4];
    float*       out       = (float*)d_out;

    const int n_in = in_sizes[0] / C_DIM;               // active input sites (100000)
    const int K    = in_sizes[1] / (C_DIM * C_DIM);     // filter volume (27)

    // 1) out[i, c] = bias[c]  (scatter-add base; also clears poisoned d_out)
    long long n4 = (long long)out_size / 4;
    bias_init_kernel<<<(unsigned)((n4 + 255) / 256), 256, 0, stream>>>(out, bias, n4);

    // 2) async gather -> WMMA GEMM -> atomic scatter
    dim3 grid((n_in + ROWS_PER_BLOCK - 1) / ROWS_PER_BLOCK, K);
    sparse_conv_wmma_kernel<<<grid, 256, 0, stream>>>(
        features, weight, rules_in, rules_out, out, n_in);
}
